// pw_GCNN_46755013984837
// MI455X (gfx1250) — compile-verified, matrix operand order checked by reading, not run on hardware
//
#include <hip/hip_runtime.h>
#include <hip/hip_bf16.h>

// ---------------------------------------------------------------------------
// GNN forward for MI455X (gfx1250, wave32, WMMA).
// Pipeline: sparse-masked-linear -> ReLU+BN -> GCNConv(WMMA GEMM + edge
// aggregate) -> ReLU+BN -> GCNConv -> ReLU+BN -> final linear.
// ---------------------------------------------------------------------------

#define NNODES 20000
#define NEDGES 640000
#define IN_F   2048
#define OUT_MASK 512
#define NCONN  8192
#define CL1    256
#define CL2    128
#define OUT_F  2
#define BN_EPS 1e-5f

typedef __attribute__((ext_vector_type(16))) _Float16 v16h;
typedef __attribute__((ext_vector_type(8)))  float    v8f;

// ---------------- utility kernels ----------------

__global__ void zero_f32(float* __restrict__ p, long long n) {
    long long i = (long long)blockIdx.x * blockDim.x + threadIdx.x;
    if (i < n) p[i] = 0.0f;
}

__global__ void fill_bias(float* __restrict__ out, const float* __restrict__ b,
                          int F, long long total) {
    long long i = (long long)blockIdx.x * blockDim.x + threadIdx.x;
    if (i < total) out[i] = b[(int)(i % F)];
}

__global__ void f32_to_f16(const float* __restrict__ in, _Float16* __restrict__ out, int n) {
    int i = blockIdx.x * blockDim.x + threadIdx.x;
    if (i < n) out[i] = (_Float16)in[i];
}

// ---------------- sparse masked linear ----------------
// h0[n, out_idx[j]] += x[n, in_idx[j]] * smw[j];  h0 initialized with sm_bias.
// One block per node: x row (8KB) and h row (2KB) resident in LDS.
__global__ __launch_bounds__(256)
void sparse_masked(const float* __restrict__ x,      // [N, IN_F]
                   const int*   __restrict__ mask,   // [NCONN, 2]
                   const float* __restrict__ smw,    // [NCONN]
                   const float* __restrict__ smb,    // [OUT_MASK]
                   float*       __restrict__ h0) {   // [N, OUT_MASK]
    __shared__ float sx[IN_F];
    __shared__ float sh[OUT_MASK];
    const int n = blockIdx.x;
    for (int i = threadIdx.x; i < IN_F; i += 256) sx[i] = x[(size_t)n * IN_F + i];
    for (int i = threadIdx.x; i < OUT_MASK; i += 256) sh[i] = smb[i];
    __syncthreads();
    for (int j = threadIdx.x; j < NCONN; j += 256) {
        int ci = mask[2 * j];
        int co = mask[2 * j + 1];
        atomicAdd(&sh[co], sx[ci] * smw[j]);          // ds_add_f32
    }
    __syncthreads();
    for (int i = threadIdx.x; i < OUT_MASK; i += 256)
        h0[(size_t)n * OUT_MASK + i] = sh[i];
}

// ---------------- BatchNorm (training-mode batch stats, biased var) --------
__global__ void bn_stats(const float* __restrict__ h, int Nn, int F,
                         int rows_per_block,
                         float* __restrict__ sums, float* __restrict__ sqs) {
    int c = blockIdx.x * blockDim.x + threadIdx.x;
    if (c >= F) return;
    int r0 = blockIdx.y * rows_per_block;
    int r1 = min(r0 + rows_per_block, Nn);
    float s = 0.f, q = 0.f;
    for (int r = r0; r < r1; ++r) {
        float v = h[(size_t)r * F + c];
        v = v > 0.f ? v : 0.f;
        s += v;
        q += v * v;
    }
    atomicAdd(&sums[c], s);
    atomicAdd(&sqs[c], q);
}

template <typename OUT_T>
__global__ void bn_apply(const float* __restrict__ h, int Nn, int F,
                         const float* __restrict__ sums, const float* __restrict__ sqs,
                         const float* __restrict__ g, const float* __restrict__ b,
                         OUT_T* __restrict__ out) {
    long long i = (long long)blockIdx.x * blockDim.x + threadIdx.x;
    if (i >= (long long)Nn * F) return;
    int c = (int)(i % F);
    float invN  = 1.0f / (float)Nn;
    float mean  = sums[c] * invN;
    float var   = sqs[c] * invN - mean * mean;
    float scale = g[c] * rsqrtf(var + BN_EPS);
    float shift = b[c] - mean * scale;
    float v = h[i];
    v = v > 0.f ? v : 0.f;
    out[i] = (OUT_T)(v * scale + shift);
}

// ---------------- GCN normalization ----------------
__global__ void compute_deg(const int* __restrict__ col, const float* __restrict__ w,
                            float* __restrict__ deg) {
    int i = blockIdx.x * blockDim.x + threadIdx.x;
    if (i < NEDGES)              atomicAdd(&deg[col[i]], w[i]);
    else if (i < NEDGES + NNODES) atomicAdd(&deg[i - NEDGES], 1.0f);  // self loop
}

__global__ void compute_norm(const int* __restrict__ row, const int* __restrict__ col,
                             const float* __restrict__ w, const float* __restrict__ deg,
                             float* __restrict__ norm) {
    int i = blockIdx.x * blockDim.x + threadIdx.x;
    if (i >= NEDGES + NNODES) return;
    int r, c; float wv;
    if (i < NEDGES) { r = row[i]; c = col[i]; wv = w[i]; }
    else            { r = c = i - NEDGES;     wv = 1.0f; }
    float dr = deg[r], dc = deg[c];
    float ir = dr > 0.f ? rsqrtf(dr) : 0.f;
    float ic = dc > 0.f ? rsqrtf(dc) : 0.f;
    norm[i] = ir * wv * ic;
}

// ---------------- WMMA GEMM: Y[M,NCOLS] = X[M,K] (f16) @ W[K,NCOLS] (f16) ----
// Block = 256 threads = 8 wave32s; one 16-row tile per block, all NCOLS cols.
// A tile staged via gfx1250 async global->LDS (ASYNCcnt). B slab staged as
// pair-packed dwords in COLUMN-major fragment layout so each lane's fragment
// is two 16B-aligned runs -> 2 x ds_load_b128 (like the A side), no VALU pack.
template <int K, int NCOLS, int TPW>
__global__ __launch_bounds__(256)
void gemm_wmma(const _Float16* __restrict__ X, const _Float16* __restrict__ W,
               float* __restrict__ Y) {
    constexpr int SBS = 20;  // dword column stride: 80B = 16B-multiple (keeps
                             // both 4-dword runs b128-aligned), gcd(20,64)=4
                             // spreads lanes over distinct bank groups.
    __shared__ alignas(16) _Float16 sA[16 * 32];
    __shared__ alignas(16) uint32_t sBf[NCOLS * SBS];  // [c][p] = {B[2p][c],B[2p+1][c]}

    const int m0   = blockIdx.x * 16;
    const int tid  = threadIdx.x;
    const int wave = tid >> 5;
    const int lane = tid & 31;
    const int row  = lane & 15;            // A row within tile
    const int kb2  = (lane < 16) ? 0 : 4;  // k-pair base per ISA 7.12.2 layout

    v8f zero = {};
    v8f acc[TPW];
#pragma unroll
    for (int t = 0; t < TPW; ++t) acc[t] = zero;

    // Per-thread A staging: one b32 (2 halves) per thread covers 16x32 tile.
    const int a_r = (tid * 2) >> 5;
    const int a_c = (tid * 2) & 31;
    const uint32_t lds_a = (uint32_t)(uintptr_t)&sA[tid * 2];  // LDS aperture: addr[31:0]

    for (int k0 = 0; k0 < K; k0 += 32) {
        __syncthreads();
        // A tile: async copy global -> LDS (gfx1250 ASYNCcnt path).
        {
            const _Float16* gp = &X[(size_t)(m0 + a_r) * K + k0 + a_c];
            asm volatile("global_load_async_to_lds_b32 %0, %1, off"
                         :: "v"(lds_a), "v"((unsigned long long)(uintptr_t)gp)
                         : "memory");
        }
        // B slab: coalesced dword reads of two consecutive W rows, pack
        // k-pairs, scatter into column-major fragment layout.
        for (int i = tid; i < 16 * (NCOLS / 2); i += 256) {
            int p  = i / (NCOLS / 2);
            int cc = (i % (NCOLS / 2)) * 2;
            uint32_t r0 = *(const uint32_t*)&W[(size_t)(k0 + 2 * p) * NCOLS + cc];
            uint32_t r1 = *(const uint32_t*)&W[(size_t)(k0 + 2 * p + 1) * NCOLS + cc];
            sBf[cc * SBS + p]       = (r0 & 0xffffu) | (r1 << 16);
            sBf[(cc + 1) * SBS + p] = (r0 >> 16)     | (r1 & 0xffff0000u);
        }
        asm volatile("s_wait_asynccnt 0x0" ::: "memory");
        __syncthreads();

        // A fragment: 2 x ds_load_b128 (pairs are row-contiguous).
        union { v16h h; uint32_t u[8]; uint4 q[2]; } af;
        af.q[0] = *(const uint4*)&sA[row * 32 + 2 * kb2];
        af.q[1] = *(const uint4*)&sA[row * 32 + 2 * kb2 + 16];
#pragma unroll
        for (int t = 0; t < TPW; ++t) {
            int col = (wave * TPW + t) * 16 + (lane & 15);
            const uint32_t* bcol = &sBf[col * SBS];
            // B fragment: 2 x ds_load_b128 from the column-major pair layout.
            union { v16h h; uint32_t u[8]; uint4 q[2]; } bf;
            bf.q[0] = *(const uint4*)&bcol[kb2];      // p = kb2 .. kb2+3
            bf.q[1] = *(const uint4*)&bcol[kb2 + 8];  // p = kb2+8 .. kb2+11
            acc[t] = __builtin_amdgcn_wmma_f32_16x16x32_f16(
                false, af.h, false, bf.h, (short)0, acc[t], false, false);
        }
    }

    // Store C: VGPR v <-> row v + (lane<16 ? 0 : 8), col = lane%16.
#pragma unroll
    for (int t = 0; t < TPW; ++t) {
        int col = (wave * TPW + t) * 16 + (lane & 15);
        int rb  = m0 + ((lane < 16) ? 0 : 8);
#pragma unroll
        for (int v = 0; v < 8; ++v)
            Y[(size_t)(rb + v) * NCOLS + col] = acc[t][v];
    }
}

// ---------------- edge aggregation: out[col] += Y[row] * norm --------------
template <int F>
__global__ void aggregate(const float* __restrict__ Y,
                          const int* __restrict__ row, const int* __restrict__ col,
                          const float* __restrict__ norm,
                          float* __restrict__ out) {
    constexpr int TPE = F / 4;  // threads per edge
    long long t = (long long)blockIdx.x * blockDim.x + threadIdx.x;
    if (t >= (long long)(NEDGES + NNODES) * TPE) return;
    int e = (int)(t / TPE);
    int f = (int)(t % TPE) * 4;
    int r, c;
    if (e < NEDGES) { r = row[e]; c = col[e]; }
    else            { r = c = e - NEDGES; }
    float nv = norm[e];
    float4 v = *(const float4*)&Y[(size_t)r * F + f];
    float* o = &out[(size_t)c * F + f];
    atomicAdd(o + 0, v.x * nv);
    atomicAdd(o + 1, v.y * nv);
    atomicAdd(o + 2, v.z * nv);
    atomicAdd(o + 3, v.w * nv);
}

// ---------------- final linear: [N,128] @ [128,2] + b ----------------------
__global__ void final_linear(const float* __restrict__ h, const float* __restrict__ W,
                             const float* __restrict__ b, float* __restrict__ out) {
    int t = blockIdx.x * blockDim.x + threadIdx.x;
    if (t >= NNODES * OUT_F) return;
    int n = t / OUT_F, o = t % OUT_F;
    float acc = b[o];
    const float* hr = &h[(size_t)n * CL2];
#pragma unroll 8
    for (int k = 0; k < CL2; ++k) acc += hr[k] * W[k * OUT_F + o];
    out[t] = acc;
}

// ---------------------------------------------------------------------------

static inline long long cdiv(long long a, long long b) { return (a + b - 1) / b; }

extern "C" void kernel_launch(void* const* d_in, const int* in_sizes, int n_in,
                              void* d_out, int out_size, void* d_ws, size_t ws_size,
                              hipStream_t stream) {
    // Inputs in setup_inputs() order (ints delivered as int32 per harness):
    const float* x       = (const float*)d_in[0];   // [N, IN_F]
    const int*   eidx    = (const int*)  d_in[1];   // [2, E]
    const float* ew      = (const float*)d_in[2];   // [E]
    const int*   smask   = (const int*)  d_in[3];   // [NCONN, 2]
    const float* smw     = (const float*)d_in[4];   // [NCONN]
    const float* smb     = (const float*)d_in[5];   // [OUT_MASK]
    const float* bns_g   = (const float*)d_in[6];
    const float* bns_b   = (const float*)d_in[7];
    const float* W1      = (const float*)d_in[8];   // [OUT_MASK, CL1]
    const float* b1      = (const float*)d_in[9];
    const float* bn1_g   = (const float*)d_in[10];
    const float* bn1_b   = (const float*)d_in[11];
    const float* W2      = (const float*)d_in[12];  // [CL1, CL2]
    const float* b2      = (const float*)d_in[13];
    const float* bn2_g   = (const float*)d_in[14];
    const float* bn2_b   = (const float*)d_in[15];
    const float* lin_W   = (const float*)d_in[16];  // [CL2, OUT_F]
    const float* lin_b   = (const float*)d_in[17];
    float* out = (float*)d_out;

    const int* e_row = eidx;            // edge_index[0]
    const int* e_col = eidx + NEDGES;   // edge_index[1]

    // Workspace carve-up (byte offsets, 128B aligned regions).
    char* ws = (char*)d_ws;
    float*    h0   = (float*)   (ws);                                  // [N,512] f32 (reused as agg2)
    _Float16* hx16 = (_Float16*)(ws + 40960000);                       // f16 GEMM input (max N*512)
    float*    y    = (float*)   (ws + 61440000);                       // GEMM out (max N*256)
    float*    agg  = (float*)   (ws + 81920000);                       // aggregate target (max N*256)
    float*    h2f  = (float*)   (ws + 102400000);                      // [N,128] f32 final features
    _Float16* W1h  = (_Float16*)(ws + 112640000);                      // [512,256] f16
    _Float16* W2h  = (_Float16*)(ws + 112902144);                      // [256,128] f16
    float*    nrm  = (float*)   (ws + 112967680);                      // [E+N]
    float*    deg  = (float*)   (ws + 115607680);                      // [N]
    float*    sums = (float*)   (ws + 115687680);                      // [512]
    float*    sqs  = (float*)   (ws + 115689728);                      // [512]

    const int TB = 256;
    const int RCHUNK = 500;   // rows per bn_stats block (40 chunks of 20000)

    // 0) f16 weight prep.
    f32_to_f16<<<cdiv(OUT_MASK * CL1, TB), TB, 0, stream>>>(W1, W1h, OUT_MASK * CL1);
    f32_to_f16<<<cdiv(CL1 * CL2, TB), TB, 0, stream>>>(W2, W2h, CL1 * CL2);

    // 0b) GCN normalization (shared by both conv layers).
    zero_f32<<<cdiv(NNODES, TB), TB, 0, stream>>>(deg, NNODES);
    compute_deg<<<cdiv(NEDGES + NNODES, TB), TB, 0, stream>>>(e_col, ew, deg);
    compute_norm<<<cdiv(NEDGES + NNODES, TB), TB, 0, stream>>>(e_row, e_col, ew, deg, nrm);

    // 1) Sparse masked linear (+ sm_bias) -> h0 [N,512].
    sparse_masked<<<NNODES, TB, 0, stream>>>(x, smask, smw, smb, h0);

    // 2) ReLU + BN -> f16 [N,512].
    zero_f32<<<cdiv(1024, TB), TB, 0, stream>>>(sums, 1024);  // sums+sqs contiguous
    bn_stats<<<dim3(2, 40), TB, 0, stream>>>(h0, NNODES, OUT_MASK, RCHUNK, sums, sqs);
    bn_apply<_Float16><<<cdiv((long long)NNODES * OUT_MASK, TB), TB, 0, stream>>>(
        h0, NNODES, OUT_MASK, sums, sqs, bns_g, bns_b, hx16);

    // 3) GCNConv1: WMMA GEMM [N,512]x[512,256], aggregate over edges, +b1.
    gemm_wmma<OUT_MASK, CL1, 2><<<NNODES / 16, TB, 0, stream>>>(hx16, W1h, y);
    fill_bias<<<cdiv((long long)NNODES * CL1, TB), TB, 0, stream>>>(
        agg, b1, CL1, (long long)NNODES * CL1);
    aggregate<CL1><<<cdiv((long long)(NEDGES + NNODES) * (CL1 / 4), TB), TB, 0, stream>>>(
        y, e_row, e_col, nrm, agg);

    // 4) ReLU + BN -> f16 [N,256].
    zero_f32<<<cdiv(1024, TB), TB, 0, stream>>>(sums, 1024);
    bn_stats<<<dim3(1, 40), TB, 0, stream>>>(agg, NNODES, CL1, RCHUNK, sums, sqs);
    bn_apply<_Float16><<<cdiv((long long)NNODES * CL1, TB), TB, 0, stream>>>(
        agg, NNODES, CL1, sums, sqs, bn1_g, bn1_b, hx16);

    // 5) GCNConv2: WMMA GEMM [N,256]x[256,128], aggregate, +b2. (h0 reused.)
    gemm_wmma<CL1, CL2, 1><<<NNODES / 16, TB, 0, stream>>>(hx16, W2h, y);
    fill_bias<<<cdiv((long long)NNODES * CL2, TB), TB, 0, stream>>>(
        h0, b2, CL2, (long long)NNODES * CL2);
    aggregate<CL2><<<cdiv((long long)(NEDGES + NNODES) * (CL2 / 4), TB), TB, 0, stream>>>(
        y, e_row, e_col, nrm, h0);

    // 6) ReLU + BN -> f32 [N,128].
    zero_f32<<<cdiv(1024, TB), TB, 0, stream>>>(sums, 1024);
    bn_stats<<<dim3(1, 40), TB, 0, stream>>>(h0, NNODES, CL2, RCHUNK, sums, sqs);
    bn_apply<float><<<cdiv((long long)NNODES * CL2, TB), TB, 0, stream>>>(
        h0, NNODES, CL2, sums, sqs, bn2_g, bn2_b, h2f);

    // 7) Final linear -> d_out [N,2].
    final_linear<<<cdiv(NNODES * OUT_F, TB), TB, 0, stream>>>(h2f, lin_W, lin_b, out);
}
